// SlidingWindowAttention_11536282157170
// MI455X (gfx1250) — matrix operand order repaired and checked by reading, not compiled, not run on hardware
//
#include <hip/hip_runtime.h>
#include <hip/hip_bf16.h>

// ---------------------------------------------------------------------------
// Sliding-window attention for MI455X (gfx1250, wave32, WMMA bf16 16x16x32).
// Pipeline:
//   1) cast x -> bf16;  transpose+cast wq/wk/wv/wo -> bf16 [n][k]
//   2) GEMM q = x@wq  (epilogue: RoPE, store bf16 [h][t][d])
//   3) GEMM k = x@wk  (epilogue: RoPE, store bf16 [kvh][t][d])
//   4) GEMM v = x@wv  (epilogue: store TRANSPOSED bf16 [kvh][d][t])
//   5) band attention per (head, 16-row q tile): 9 key tiles of 16,
//      S via WMMA -> LDS f32 -> softmax -> bf16 P (padded K=160) -> P@V WMMA
//   6) GEMM out = ab@wo (f32 store to d_out)
// Whole bf16 working set (~76 MB) fits in the 192 MB L2, so GEMMs stream
// fragments straight from global (L2-resident) with prefetch hints.
// GEMM wave tile is 32x64 (2 A-frags x 4 B-frags = 8 WMMA per k-step) for
// ~21 flops/byte of L2 traffic; block tile 128x128 (8 waves, 4x2).
// ---------------------------------------------------------------------------

typedef __attribute__((ext_vector_type(16))) __bf16 v16bf;
typedef __attribute__((ext_vector_type(8)))  float  v8f;

union Frag16 { uint4 u[2]; v16bf v; };

#define WMMA_BF16(Afr, Bfr, Cfr) \
  __builtin_amdgcn_wmma_f32_16x16x32_bf16(false, (Afr), false, (Bfr), (short)0, (Cfr), false, false)

// A-matrix fragment (16x32 bf16, row-major source, stride in elements).
// lane<16 : K = kb..kb+15; lane>=16: K = kb+16..kb+31.
__device__ __forceinline__ v16bf load_a_frag(const __bf16* base, int stride,
                                             int row16, int kb, int lane) {
  const int lr = lane & 15, koff = (lane >> 4) * 16;
  const __bf16* p = base + (size_t)(row16 + lr) * stride + kb + koff;
  Frag16 f;
  f.u[0] = *(const uint4*)p;
  f.u[1] = *(const uint4*)(p + 8);
  return f.v;
}

// B-matrix fragment (32x16 bf16). Source is B^T row-major ([n][k]).
// lanes 0-15 hold K=kb..kb+15 of column lr; lanes 16-31 hold K=kb+16..kb+31.
__device__ __forceinline__ v16bf load_b_frag(const __bf16* baseT, int stride,
                                             int col16, int kb, int lane) {
  const int lr = lane & 15, koff = (lane >> 4) * 16;
  const __bf16* p = baseT + (size_t)(col16 + lr) * stride + kb + koff;
  Frag16 f;
  f.u[0] = *(const uint4*)p;
  f.u[1] = *(const uint4*)(p + 8);
  return f.v;
}

// --------------------------- elementwise casts -----------------------------

__global__ __launch_bounds__(256) void cvt_f32_to_bf16_x4(const float* __restrict__ x,
                                                          __bf16* __restrict__ y, int n4) {
  int i = blockIdx.x * blockDim.x + threadIdx.x;
  if (i >= n4) return;
  float4 v = ((const float4*)x)[i];
  union { __bf16 h[4]; uint2 u; } p;
  p.h[0] = (__bf16)v.x; p.h[1] = (__bf16)v.y;
  p.h[2] = (__bf16)v.z; p.h[3] = (__bf16)v.w;
  ((uint2*)y)[i] = p.u;
}

// W (K x N, f32, row-major) -> Wt (N x K, bf16, row-major)
__global__ __launch_bounds__(256) void transpose_to_bf16(const float* __restrict__ W,
                                                         __bf16* __restrict__ Wt,
                                                         int K, int N) {
  __shared__ float tile[32][33];
  const int k0 = blockIdx.x * 32, n0 = blockIdx.y * 32;
  const int tx = threadIdx.x, ty = threadIdx.y;   // 32 x 8
#pragma unroll
  for (int i = 0; i < 32; i += 8)
    tile[ty + i][tx] = W[(size_t)(k0 + ty + i) * N + n0 + tx];
  __syncthreads();
#pragma unroll
  for (int i = 0; i < 32; i += 8)
    Wt[(size_t)(n0 + ty + i) * K + k0 + tx] = (__bf16)tile[tx][ty + i];
}

// ------------------------------- GEMM --------------------------------------
// C(MxN) = A(MxK,bf16) x B(KxN) with B given as Bt(NxK,bf16).
// Block: 256 thr = 8 waves (4x2), block tile 128(M) x 128(N);
// wave tile 32x64 = 2 A-frags x 4 B-frags -> 8 WMMA per 32-deep k-step.
// mode 0: RoPE epilogue, store bf16 to dbf[[col>>7]][row][col&127] (head-major)
// mode 1: store bf16 TRANSPOSED to dbf[[col>>7]][col&127][row]
// mode 2: store f32 to dfl[row*N + col]

__global__ __launch_bounds__(256) void gemm_bf16(const __bf16* __restrict__ A,
                                                 const __bf16* __restrict__ Bt,
                                                 int M, int N, int K, int mode,
                                                 __bf16* __restrict__ dbf,
                                                 float* __restrict__ dfl) {
  const int lane = threadIdx.x & 31;
  const int wid  = threadIdx.x >> 5;
  const int wm = wid >> 1, wn = wid & 1;          // 4 x 2 wave grid
  const int gm = blockIdx.x * 128 + wm * 32;
  const int gn = blockIdx.y * 128 + wn * 64;

  v8f acc[2][4];
#pragma unroll
  for (int i = 0; i < 2; ++i)
#pragma unroll
    for (int j = 0; j < 4; ++j)
      acc[i][j] = (v8f){0.f, 0.f, 0.f, 0.f, 0.f, 0.f, 0.f, 0.f};

  for (int kb = 0; kb < K; kb += 32) {
    v16bf a0 = load_a_frag(A, K, gm,      kb, lane);
    v16bf a1 = load_a_frag(A, K, gm + 16, kb, lane);
    v16bf b0 = load_b_frag(Bt, K, gn,      kb, lane);
    v16bf b1 = load_b_frag(Bt, K, gn + 16, kb, lane);
    v16bf b2 = load_b_frag(Bt, K, gn + 32, kb, lane);
    v16bf b3 = load_b_frag(Bt, K, gn + 48, kb, lane);
    // L2 prefetch a few k-tiles ahead (emits global_prefetch_b8).
    if (kb + 96 < K) {
      __builtin_prefetch(A  + (size_t)(gm + (lane & 15)) * K + kb + 96, 0, 1);
      __builtin_prefetch(Bt + (size_t)(gn + (lane & 15)) * K + kb + 96, 0, 1);
    }
    acc[0][0] = WMMA_BF16(a0, b0, acc[0][0]);
    acc[0][1] = WMMA_BF16(a0, b1, acc[0][1]);
    acc[0][2] = WMMA_BF16(a0, b2, acc[0][2]);
    acc[0][3] = WMMA_BF16(a0, b3, acc[0][3]);
    acc[1][0] = WMMA_BF16(a1, b0, acc[1][0]);
    acc[1][1] = WMMA_BF16(a1, b1, acc[1][1]);
    acc[1][2] = WMMA_BF16(a1, b2, acc[1][2]);
    acc[1][3] = WMMA_BF16(a1, b3, acc[1][3]);
  }

  const int lr = lane & 15, hi = lane >> 4;
  const float kNegLn10k_64 = -0.14391156f;   // -ln(10000)/64

#pragma unroll
  for (int fm = 0; fm < 2; ++fm) {
#pragma unroll
    for (int fn = 0; fn < 4; ++fn) {
      const int col = gn + fn * 16 + lr;
      if (mode == 0) {
        // RoPE: even col holds xr, odd col holds xi; pair is adjacent lane.
        const int h = col >> 7, d = col & 127;
        const float invf = __expf((float)(d >> 1) * kNegLn10k_64);
        const bool odd = (col & 1) != 0;
#pragma unroll
        for (int r = 0; r < 8; ++r) {
          const int row = gm + fm * 16 + r + 8 * hi;
          float v = acc[fm][fn][r];
          float pv = __shfl_xor(v, 1);
          float s, c;
          __sincosf((float)row * invf, &s, &c);
          float res = odd ? fmaf(pv, s, v * c) : fmaf(v, c, -pv * s);
          dbf[(size_t)h * M * 128 + (size_t)row * 128 + d] = (__bf16)res;
        }
      } else if (mode == 1) {
        // Transposed store: lane's 8 rows are consecutive -> one 16B store.
        const int h = col >> 7, d = col & 127;
        const int row0 = gm + fm * 16 + 8 * hi;
        union { __bf16 h8[8]; uint4 u; } pk;
#pragma unroll
        for (int r = 0; r < 8; ++r) pk.h8[r] = (__bf16)acc[fm][fn][r];
        *(uint4*)(dbf + (size_t)h * 128 * M + (size_t)d * M + row0) = pk.u;
      } else {
#pragma unroll
        for (int r = 0; r < 8; ++r) {
          const int row = gm + fm * 16 + r + 8 * hi;
          dfl[(size_t)row * N + col] = acc[fm][fn][r];
        }
      }
    }
  }
}

// ----------------------------- attention -----------------------------------
// One wave per (head, 16-row q tile). Band |q-k| <= 64 -> exactly 9 key
// tiles of 16 (kb = q0-64 .. q0+80). hd=128 -> 4 WMMA k-steps for S.
// P padded to K=160 (5 k-steps) for P@V; invalid/pad keys have P=0 and
// their V addresses are clamped in-range.

__global__ __launch_bounds__(32) void attn_kernel(const __bf16* __restrict__ qh,
                                                  const __bf16* __restrict__ kh,
                                                  const __bf16* __restrict__ vt,
                                                  const float* __restrict__ sink,
                                                  __bf16* __restrict__ ab) {
  constexpr int T = 4096, HD = 128;
  const int lane = threadIdx.x;
  const int h = blockIdx.y;
  const int q0 = blockIdx.x * 16;
  const int kvh = h >> 2;                 // GQA: 4 q-heads per kv-head
  const int lr = lane & 15, hi = lane >> 4;
  const int koff_a = hi * 16, koff_b = hi * 16;

  __shared__ __align__(16) float  sS[16][160];
  __shared__ float sSum[16];
  __shared__ __align__(16) __bf16 sP[16][160];

  // ---- load q fragments (A-matrix role), 4 k-steps over hd=128 ----
  v16bf QA[4];
  {
    const __bf16* qb = qh + (size_t)h * T * HD + (size_t)(q0 + lr) * HD;
#pragma unroll
    for (int ks = 0; ks < 4; ++ks) {
      Frag16 f;
      const __bf16* p = qb + ks * 32 + koff_a;
      f.u[0] = *(const uint4*)p;
      f.u[1] = *(const uint4*)(p + 8);
      QA[ks] = f.v;
    }
  }

  const float scale = 0.08838834764831845f;   // 128^-0.5
  const float snk = sink[h];
  const __bf16* kbp = kh + (size_t)kvh * T * HD;

  // ---- S = q k^T over 9 key tiles, masked, -> LDS ----
  for (int j = 0; j < 9; ++j) {
    const int kb = q0 - 64 + j * 16;
    const int key = kb + lr;                       // this lane's B column
    const bool kvalid = (key >= 0) && (key < T);
    v8f acc = (v8f){0.f, 0.f, 0.f, 0.f, 0.f, 0.f, 0.f, 0.f};
#pragma unroll
    for (int ks = 0; ks < 4; ++ks) {
      Frag16 f;
      if (kvalid) {
        const __bf16* p = kbp + (size_t)key * HD + ks * 32 + koff_b;
        f.u[0] = *(const uint4*)p;
        f.u[1] = *(const uint4*)(p + 8);
      } else {
        f.u[0] = uint4{0, 0, 0, 0};
        f.u[1] = uint4{0, 0, 0, 0};
      }
      acc = WMMA_BF16(QA[ks], f.v, acc);
    }
#pragma unroll
    for (int r = 0; r < 8; ++r) {
      const int m = r + 8 * hi;
      const int dqk = (q0 + m) - key;
      const bool valid = kvalid && (dqk <= 64) && (dqk >= -64);
      sS[m][j * 16 + lr] = valid ? fmaf(acc[r], scale, snk) : -INFINITY;
    }
  }
  __syncthreads();

  // ---- softmax (unnormalized P in bf16; row sums kept for final scale) ----
  if (lane < 16) {
    const int m = lane;
    float mx = -INFINITY;
    for (int kk = 0; kk < 144; ++kk) mx = fmaxf(mx, sS[m][kk]);
    float sum = 0.f;
    for (int kk = 0; kk < 144; ++kk) {
      float p = __expf(sS[m][kk] - mx);
      sum += p;
      sP[m][kk] = (__bf16)p;
    }
    sSum[m] = sum;                       // >= 1 (diagonal always in band)
  } else {
    const int m = lane - 16;
    for (int kk = 144; kk < 160; ++kk) sP[m][kk] = (__bf16)0.f;
  }
  __syncthreads();

  // ---- out(16x128) = P(16x160) @ V(160x128); V read from vt[kvh][d][t] ----
  v8f o[8];
#pragma unroll
  for (int fn = 0; fn < 8; ++fn) o[fn] = (v8f){0.f, 0.f, 0.f, 0.f, 0.f, 0.f, 0.f, 0.f};

  for (int kt = 0; kt < 5; ++kt) {
    const int kbase = kt * 32;
    Frag16 pa;
    {
      const __bf16* pp = &sP[lr][kbase + koff_a];
      pa.u[0] = *(const uint4*)pp;
      pa.u[1] = *(const uint4*)(pp + 8);
    }
    const int t0 = q0 - 64 + kbase + koff_b;
    const int t0c = min(max(t0, 0), T - 8);        // clamp: masked keys have P=0
    const int t1c = min(max(t0 + 8, 0), T - 8);
#pragma unroll
    for (int fn = 0; fn < 8; ++fn) {
      const int d = fn * 16 + lr;
      const __bf16* vb = vt + (size_t)kvh * HD * T + (size_t)d * T;
      Frag16 f;
      f.u[0] = *(const uint4*)(vb + t0c);
      f.u[1] = *(const uint4*)(vb + t1c);
      o[fn] = WMMA_BF16(pa.v, f.v, o[fn]);
    }
  }

  float rs[8];
#pragma unroll
  for (int r = 0; r < 8; ++r) rs[r] = 1.0f / sSum[r + 8 * hi];

#pragma unroll
  for (int fn = 0; fn < 8; ++fn) {
    const int c = h * HD + fn * 16 + lr;
#pragma unroll
    for (int r = 0; r < 8; ++r) {
      const int t = q0 + r + 8 * hi;
      ab[(size_t)t * 2048 + c] = (__bf16)(o[fn][r] * rs[r]);
    }
  }
}

// ------------------------------- launch ------------------------------------

extern "C" void kernel_launch(void* const* d_in, const int* in_sizes, int n_in,
                              void* d_out, int out_size, void* d_ws, size_t ws_size,
                              hipStream_t stream) {
  (void)in_sizes; (void)n_in; (void)out_size; (void)ws_size;
  constexpr int T = 4096, C = 2048, KVC = 512, HD = 128, NH = 16, NKV = 4;

  const float* x    = (const float*)d_in[0];
  const float* wq   = (const float*)d_in[1];
  const float* wk   = (const float*)d_in[2];
  const float* wv   = (const float*)d_in[3];
  const float* wo   = (const float*)d_in[4];
  const float* sink = (const float*)d_in[5];
  float* out = (float*)d_out;

  char* ws = (char*)d_ws;
  size_t off = 0;
  __bf16* xb  = (__bf16*)(ws + off); off += (size_t)T * C * 2;        // 16 MB
  __bf16* wqT = (__bf16*)(ws + off); off += (size_t)C * C * 2;        //  8 MB
  __bf16* wkT = (__bf16*)(ws + off); off += (size_t)KVC * C * 2;      //  2 MB
  __bf16* wvT = (__bf16*)(ws + off); off += (size_t)KVC * C * 2;      //  2 MB
  __bf16* woT = (__bf16*)(ws + off); off += (size_t)C * C * 2;        //  8 MB
  __bf16* qh  = (__bf16*)(ws + off); off += (size_t)NH * T * HD * 2;  // 16 MB
  __bf16* kh  = (__bf16*)(ws + off); off += (size_t)NKV * T * HD * 2; //  4 MB
  __bf16* vt  = (__bf16*)(ws + off); off += (size_t)NKV * HD * T * 2; //  4 MB
  __bf16* ab  = (__bf16*)(ws + off); off += (size_t)T * C * 2;        // 16 MB
  // total ~76 MB of workspace

  // 1) casts / transposes
  {
    int n4 = T * C / 4;
    cvt_f32_to_bf16_x4<<<(n4 + 255) / 256, 256, 0, stream>>>(x, xb, n4);
    dim3 blk(32, 8);
    transpose_to_bf16<<<dim3(C / 32, C / 32),   blk, 0, stream>>>(wq, wqT, C, C);
    transpose_to_bf16<<<dim3(C / 32, KVC / 32), blk, 0, stream>>>(wk, wkT, C, KVC);
    transpose_to_bf16<<<dim3(C / 32, KVC / 32), blk, 0, stream>>>(wv, wvT, C, KVC);
    transpose_to_bf16<<<dim3(C / 32, C / 32),   blk, 0, stream>>>(wo, woT, C, C);
  }

  // 2) q/k (RoPE epilogue) and v (transposed-store epilogue)
  gemm_bf16<<<dim3(T / 128, C / 128),   256, 0, stream>>>(xb, wqT, T, C,   C, 0, qh, nullptr);
  gemm_bf16<<<dim3(T / 128, KVC / 128), 256, 0, stream>>>(xb, wkT, T, KVC, C, 0, kh, nullptr);
  gemm_bf16<<<dim3(T / 128, KVC / 128), 256, 0, stream>>>(xb, wvT, T, KVC, C, 1, vt, nullptr);

  // 3) banded attention: one wave per (16-row q tile, head)
  attn_kernel<<<dim3(T / 16, NH), 32, 0, stream>>>(qh, kh, vt, sink, ab);

  // 4) output projection (f32 store)
  gemm_bf16<<<dim3(T / 128, C / 128), 256, 0, stream>>>(ab, woT, T, C, C, 2, nullptr, out);
}
